// AWDTransformerXL_52793738002679
// MI455X (gfx1250) — compile-verified, matrix operand order checked by reading, not run on hardware
//
#include <hip/hip_runtime.h>
#include <cstdint>
#include <cstddef>

// ---------------- CDNA5 WMMA types/helpers ----------------
typedef __attribute__((ext_vector_type(16))) __bf16 v16bf;
typedef __attribute__((ext_vector_type(8)))  float  v8f;

union Frag16 { v16bf v; uint4 q[2]; unsigned short u[16]; };
union Acc8   { v8f v; float f[8]; };

__device__ inline unsigned short f2bf(float f) {
  unsigned int u = __float_as_uint(f);
  u += 0x7FFFu + ((u >> 16) & 1u);          // round-to-nearest-even
  return (unsigned short)(u >> 16);
}

__device__ inline v8f wmma_bf16(v16bf a, v16bf b, v8f c) {
  // D = A(16x32 bf16) x B(32x16 bf16) + C(16x16 f32)
  return __builtin_amdgcn_wmma_f32_16x16x32_bf16(false, a, false, b, (short)0, c,
                                                 false, false);
}

// Load one 16x32 bf16 fragment (K-major rows) from LDS.
// ISA layout: lanes 0-15 hold K={0..7,16..23}, lanes 16-31 hold K={8..15,24..31}.
__device__ inline v16bf lds_frag(const unsigned short* buf, int row, int stride, int lane) {
  Frag16 f;
  int off = (lane < 16) ? 0 : 8;
  const unsigned short* p = buf + row * stride;
  f.q[0] = *(const uint4*)(p + off);
  f.q[1] = *(const uint4*)(p + 16 + off);
  return f.v;
}

// CDNA5 async global->LDS copy (no VGPR transit, tracked by ASYNCcnt).
// lds_byte_off = offset within the workgroup LDS allocation (generic-ptr low 32b).
__device__ inline void async_b128(unsigned lds_byte_off, const void* g) {
  asm volatile("global_load_async_to_lds_b128 %0, %1, off"
               :: "v"(lds_byte_off), "v"((unsigned long long)(size_t)g)
               : "memory");
}
__device__ inline void wait_async0() {
  asm volatile("s_wait_asynccnt 0" ::: "memory");
}

// ---------------- model dims ----------------
#define QL 1024
#define BB 4
#define VV 32000
#define LL 4
#define NHH 16
#define DHH 64
#define DMM 1024
#define DII 4096
#define MM  (QL * BB)   // 4096 token rows

// ---------------- bulk fp32 -> bf16 conversion (one pass per tensor) --------
__global__ __launch_bounds__(256) void cvt_bf16_kernel(
    const float* __restrict__ src, unsigned short* __restrict__ dst, int n)
{
  int i = (blockIdx.x * 256 + threadIdx.x) * 4;
  if (i < n) {
    float4 f = *(const float4*)(src + i);
    union { unsigned short u[4]; uint2 q; } t;
    t.u[0] = f2bf(f.x); t.u[1] = f2bf(f.y); t.u[2] = f2bf(f.z); t.u[3] = f2bf(f.w);
    *(uint2*)(dst + i) = t.q;
  }
}

// ---------------- embedding ----------------
__global__ __launch_bounds__(256) void embed_kernel(
    const int* __restrict__ tok, const float* __restrict__ emb,
    float* __restrict__ hf, unsigned short* __restrict__ hb)
{
  int m = blockIdx.x;                 // m = q*B + b (matches dec_inp flat order)
  int t = tok[m];
  const float* row = emb + (size_t)t * DMM;
  for (int d = threadIdx.x; d < DMM; d += 256) {
    float v = row[d] * 32.0f;         // sqrt(1024)
    hf[(size_t)m * DMM + d] = v;
    hb[(size_t)m * DMM + d] = f2bf(v);
  }
}

// -------- GEMM: out[M,N] = A_bf16[M,K] * W_bf16[N,K]^T, double-buffered -----
// 256 threads = 8 waves; tile 128x128x32; each wave does 64x32 via 4x2 WMMA.
// Staging uses GLOBAL_LOAD_ASYNC_TO_LDS_B128: global->LDS with zero VGPR
// transit (ASYNCcnt).  Per step: wait own copies (buf[cur] complete), barrier
// (publishes tile; also proves step-1 readers of buf[1-cur] are done), issue
// next tile's async copies into buf[1-cur], then 8 WMMAs from buf[cur] while
// the copies fly.
template <bool STORE_BF>
__global__ __attribute__((amdgpu_waves_per_eu(1, 8))) __launch_bounds__(256)
void gemm_bf16_kernel(
    const unsigned short* __restrict__ A, const unsigned short* __restrict__ Wb,
    void* __restrict__ out, int M, int N, int K)
{
  __shared__ unsigned short As[2][128 * 40];   // rows padded to 40 (80B, 16B-aligned)
  __shared__ unsigned short Bs[2][128 * 40];

  const int tid  = threadIdx.x;
  const int lane = tid & 31;
  const int wv   = tid >> 5;
  const int wm   = wv >> 2;                 // 0..1 : 64-row slab
  const int wn   = wv & 3;                  // 0..3 : 32-col slab
  const int mbase = blockIdx.y * 128;
  const int nbase = blockIdx.x * 128;
  const int lm    = lane & 15;
  const int half8 = (lane < 16) ? 0 : 8;

  Acc8 c[4][2];
  for (int i = 0; i < 4; ++i)
    for (int j = 0; j < 2; ++j) { v8f z = {0,0,0,0,0,0,0,0}; c[i][j].v = z; }

  const int srow  = tid >> 1;               // staging: 128 rows x 2 halves of 16
  const int shalf = tid & 1;
  const size_t aBase = (size_t)(mbase + srow) * K + shalf * 16;
  const size_t bBase = (size_t)(nbase + srow) * K + shalf * 16;
  const unsigned aOff0 = (unsigned)(size_t)&As[0][srow * 40 + shalf * 16];
  const unsigned aOff1 = (unsigned)(size_t)&As[1][srow * 40 + shalf * 16];
  const unsigned bOff0 = (unsigned)(size_t)&Bs[0][srow * 40 + shalf * 16];
  const unsigned bOff1 = (unsigned)(size_t)&Bs[1][srow * 40 + shalf * 16];

  const int nsteps = K >> 5;

  { // prologue: async-copy tile 0 into buf0
    const unsigned short* gA = A + aBase;
    const unsigned short* gB = Wb + bBase;
    async_b128(aOff0, gA); async_b128(aOff0 + 16, gA + 8);
    async_b128(bOff0, gB); async_b128(bOff0 + 16, gB + 8);
  }

  for (int step = 0; step < nsteps; ++step) {
    const int cur = step & 1;
    wait_async0();                         // own copies into buf[cur] landed
    __syncthreads();                       // everyone's copies landed
    if (step + 1 < nsteps) {               // next tile -> back buffer, overlaps WMMAs
      const unsigned short* gA = A + aBase + (size_t)(step + 1) * 32;
      const unsigned short* gB = Wb + bBase + (size_t)(step + 1) * 32;
      const unsigned ao = cur ? aOff0 : aOff1;
      const unsigned bo = cur ? bOff0 : bOff1;
      async_b128(ao, gA); async_b128(ao + 16, gA + 8);
      async_b128(bo, gB); async_b128(bo + 16, gB + 8);
      if (step + 2 < nsteps) {
        __builtin_prefetch(A + aBase + (size_t)(step + 2) * 32, 0, 1);
        __builtin_prefetch(Wb + bBase + (size_t)(step + 2) * 32, 0, 1);
      }
    }
    {
      v16bf a[4], b[2];
      for (int i = 0; i < 4; ++i)
        a[i] = lds_frag(&As[cur][0], wm * 64 + i * 16 + lm, 40, lane);
      for (int j = 0; j < 2; ++j)
        b[j] = lds_frag(&Bs[cur][0], wn * 32 + j * 16 + lm, 40, lane);
      for (int i = 0; i < 4; ++i)
        for (int j = 0; j < 2; ++j)
          c[i][j].v = wmma_bf16(a[i], b[j], c[i][j].v);
    }
  }

  // epilogue: incremental addressing, single output format per instantiation
  const int colBase = nbase + wn * 32 + lm;
  const int rowBase = mbase + wm * 64 + half8;
  if constexpr (STORE_BF) {
    unsigned short* o = (unsigned short*)out;
    for (int i = 0; i < 4; ++i) {
      size_t idx = (size_t)(rowBase + i * 16) * N + colBase;
      for (int r = 0; r < 8; ++r) {
        o[idx]      = f2bf(c[i][0].f[r]);
        o[idx + 16] = f2bf(c[i][1].f[r]);
        idx += N;
      }
    }
  } else {
    float* o = (float*)out;
    for (int i = 0; i < 4; ++i) {
      size_t idx = (size_t)(rowBase + i * 16) * N + colBase;
      for (int r = 0; r < 8; ++r) {
        o[idx]      = c[i][0].f[r];
        o[idx + 16] = c[i][1].f[r];
        idx += N;
      }
    }
  }
}

// ---------------- causal flash attention per (batch, head) ----------------
// grid = (16 q-tiles of 64 rows, 64 = b*16+h), block = 128 (4 waves, 16 q-rows each)
__global__ __launch_bounds__(128) void attn_kernel(
    const unsigned short* __restrict__ qkv,   // [M, 3072] bf16 : [Q|K|V] x head*64+dh
    unsigned short* __restrict__ att)         // [M, 1024] bf16
{
  __shared__ unsigned short Vt[64 * 40];      // V^T tile: [dh 0..63][key 0..31]
  __shared__ unsigned short Pl[4][16 * 40];   // per-wave P tile 16x32

  const int tid  = threadIdx.x;
  const int lane = tid & 31;
  const int wv   = tid >> 5;
  const int lm   = lane & 15;
  const int half8 = (lane < 16) ? 0 : 8;
  const int off   = (lane < 16) ? 0 : 8;

  const int qtile = blockIdx.x;
  const int bh    = blockIdx.y;
  const int b     = bh >> 4;
  const int h     = bh & 15;
  const int qbase = qtile * 64;
  const int wqb   = qbase + wv * 16;
  const int wqmax = wqb + 15;

  // Q A-fragments (16 rows x 64 dh = two k=32 fragments), straight from global
  v16bf qa0, qa1;
  {
    int qrow = wqb + lm;
    const unsigned short* qp = qkv + (size_t)(qrow * BB + b) * 3072 + h * 64;
    Frag16 f0, f1;
    f0.q[0] = *(const uint4*)(qp + off);      f0.q[1] = *(const uint4*)(qp + 16 + off);
    f1.q[0] = *(const uint4*)(qp + 32 + off); f1.q[1] = *(const uint4*)(qp + 48 + off);
    qa0 = f0.v; qa1 = f1.v;
  }

  Acc8 o[4];
  for (int i = 0; i < 4; ++i) { v8f z = {0,0,0,0,0,0,0,0}; o[i].v = z; }
  float rmax[8], rsum[8];
  for (int r = 0; r < 8; ++r) { rmax[r] = -1e30f; rsum[r] = 0.f; }

  const int nkt = (qbase + 63) / 32 + 1;      // causal key-tile count for this block
  for (int kt = 0; kt < nkt; ++kt) {
    const int kbase = kt * 32;
    __syncthreads();
    { // cooperatively stage V^T (32 keys x 64 dh)
      int ki  = tid >> 2;                     // 0..31 key in tile
      int dhb = (tid & 3) * 16;               // 0,16,32,48
      const unsigned short* vp =
          qkv + (size_t)((kbase + ki) * BB + b) * 3072 + 2048 + h * 64 + dhb;
      Frag16 t;
      t.q[0] = ((const uint4*)vp)[0];
      t.q[1] = ((const uint4*)vp)[1];
      for (int j = 0; j < 16; ++j) Vt[(dhb + j) * 40 + ki] = t.u[j];
    }
    __syncthreads();
    if (kbase > wqmax) continue;              // fully masked for this wave

    // S = Q K^T over dh=64 : 2 n-tiles x 2 k-fragments
    Acc8 sc[2];
    for (int nt = 0; nt < 2; ++nt) {
      int key = kbase + nt * 16 + lm;
      const unsigned short* kp = qkv + (size_t)(key * BB + b) * 3072 + 1024 + h * 64;
      Frag16 kb0, kb1;
      kb0.q[0] = *(const uint4*)(kp + off);      kb0.q[1] = *(const uint4*)(kp + 16 + off);
      kb1.q[0] = *(const uint4*)(kp + 32 + off); kb1.q[1] = *(const uint4*)(kp + 48 + off);
      v8f z = {0,0,0,0,0,0,0,0};
      z = wmma_bf16(qa0, kb0.v, z);
      z = wmma_bf16(qa1, kb1.v, z);
      sc[nt].v = z;
    }

    // online softmax + P write (16-lane-half reductions match C-fragment layout)
    unsigned short* pl = &Pl[wv][0];
    for (int r = 0; r < 8; ++r) {
      int qrow = wqb + r + half8;
      float v0 = sc[0].f[r] * 0.125f;
      float v1 = sc[1].f[r] * 0.125f;
      if (kbase + lm      > qrow) v0 = -1e30f;
      if (kbase + 16 + lm > qrow) v1 = -1e30f;
      float m2 = fmaxf(v0, v1);
      for (int msk = 8; msk; msk >>= 1) m2 = fmaxf(m2, __shfl_xor(m2, msk, 32));
      float nm   = fmaxf(rmax[r], m2);
      float corr = __expf(rmax[r] - nm);
      rmax[r] = nm;
      float e0 = __expf(v0 - nm), e1 = __expf(v1 - nm);
      float es = e0 + e1;
      for (int msk = 8; msk; msk >>= 1) es += __shfl_xor(es, msk, 32);
      rsum[r] = rsum[r] * corr + es;
      for (int c4 = 0; c4 < 4; ++c4) o[c4].f[r] *= corr;
      pl[(r + half8) * 40 + lm]      = f2bf(e0);
      pl[(r + half8) * 40 + 16 + lm] = f2bf(e1);
    }
    // cross-lane DS RAW inside one wave: make ordering explicit
    asm volatile("s_wait_dscnt 0" ::: "memory");

    v16bf pa = lds_frag(pl, lm, 40, lane);    // P as 16x32 A fragment
    for (int c4 = 0; c4 < 4; ++c4) {
      v16bf vb = lds_frag(Vt, c4 * 16 + lm, 40, lane);
      o[c4].v = wmma_bf16(pa, vb, o[c4].v);
    }
  }

  for (int c4 = 0; c4 < 4; ++c4)
    for (int r = 0; r < 8; ++r) {
      int qrow = wqb + r + half8;
      size_t mo = (size_t)(qrow * BB + b) * DMM + h * 64 + c4 * 16 + lm;
      att[mo] = f2bf(o[c4].f[r] / rsum[r]);
    }
}

// ---------------- residual + layernorm (fp32 master, bf16 copy) ----------------
__global__ __launch_bounds__(256) void ln_kernel(
    const float* resid, const float* __restrict__ delta,
    const float* __restrict__ g, const float* __restrict__ bta,
    float* hf, unsigned short* __restrict__ hb)
{
  __shared__ float red[8];
  const int m = blockIdx.x, tid = threadIdx.x, lane = tid & 31, wv = tid >> 5;
  const float* rp = resid + (size_t)m * DMM;
  const float* dp = delta + (size_t)m * DMM;
  float x[4]; float s = 0.f;
  for (int i = 0; i < 4; ++i) { int idx = tid + i * 256; x[i] = rp[idx] + dp[idx]; s += x[i]; }
  for (int msk = 16; msk; msk >>= 1) s += __shfl_xor(s, msk, 32);
  if (lane == 0) red[wv] = s;
  __syncthreads();
  float tot = 0.f; for (int i = 0; i < 8; ++i) tot += red[i];
  float mu = tot * (1.f / DMM);
  float vs = 0.f;
  for (int i = 0; i < 4; ++i) { float d = x[i] - mu; vs += d * d; }
  for (int msk = 16; msk; msk >>= 1) vs += __shfl_xor(vs, msk, 32);
  __syncthreads();
  if (lane == 0) red[wv] = vs;
  __syncthreads();
  float vtot = 0.f; for (int i = 0; i < 8; ++i) vtot += red[i];
  float rstd = rsqrtf(vtot * (1.f / DMM) + 1e-5f);
  for (int i = 0; i < 4; ++i) {
    int idx = tid + i * 256;
    float y = (x[i] - mu) * rstd * g[idx] + bta[idx];
    hf[(size_t)m * DMM + idx] = y;
    hb[(size_t)m * DMM + idx] = f2bf(y);
  }
}

// ---------------- streaming tied-logits + log-softmax loss ----------------
// grid = 128 blocks of 32 token rows; 4 waves split the vocab; logits never stored.
// bf16 embedding (65 MB) stays resident in the 192 MB L2 across the sweep.
__global__ __launch_bounds__(128) void loss_kernel(
    const unsigned short* __restrict__ hb, const unsigned short* __restrict__ embb,
    const float* __restrict__ bias, const int* __restrict__ target,
    float* __restrict__ loss)
{
  __shared__ unsigned short Hs[32 * 1032];
  __shared__ float wmaxS[4][32], wsumS[4][32], wtgtS[4][32];
  const int tid = threadIdx.x, lane = tid & 31, wv = tid >> 5;
  const int lm = lane & 15, half8 = (lane < 16) ? 0 : 8, off = (lane < 16) ? 0 : 8;
  const int rb = blockIdx.x * 32;

  for (int idx = tid; idx < 32 * 64; idx += 128) {   // 64 chunks of 16 bf16 per row
    int r = idx >> 6, cchunk = idx & 63;
    const uint4* g = (const uint4*)(hb + (size_t)(rb + r) * DMM + cchunk * 16);
    uint4 a0 = g[0], a1 = g[1];
    *(uint4*)&Hs[r * 1032 + cchunk * 16]     = a0;
    *(uint4*)&Hs[r * 1032 + cchunk * 16 + 8] = a1;
  }
  int tgt[2][8];
  float rm[2][8], rs[2][8], rt[2][8];
  for (int t2 = 0; t2 < 2; ++t2)
    for (int r = 0; r < 8; ++r) {
      tgt[t2][r] = target[rb + t2 * 16 + r + half8];
      rm[t2][r] = -1e30f; rs[t2][r] = 0.f; rt[t2][r] = -1e30f;
    }
  __syncthreads();

  for (int nt = wv; nt < VV / 16; nt += 4) {         // vocab n-tiles, 16 wide
    const int v = nt * 16 + lm;
    Frag16 fb;                                       // emb row (pre-converted bf16)
    const unsigned short* ep = embb + (size_t)v * DMM;
    Acc8 c[2];
    { v8f z = {0,0,0,0,0,0,0,0}; c[0].v = z; c[1].v = z; }
    for (int kk = 0; kk < DMM; kk += 32) {
      fb.q[0] = *(const uint4*)(ep + kk + off);
      fb.q[1] = *(const uint4*)(ep + kk + 16 + off);
      for (int t2 = 0; t2 < 2; ++t2) {
        Frag16 fa;
        const unsigned short* hp = &Hs[(t2 * 16 + lm) * 1032 + kk];
        fa.q[0] = *(const uint4*)(hp + off);
        fa.q[1] = *(const uint4*)(hp + 16 + off);
        c[t2].v = wmma_bf16(fa.v, fb.v, c[t2].v);
      }
    }
    float bia = bias[v];
    for (int t2 = 0; t2 < 2; ++t2)
      for (int r = 0; r < 8; ++r) {
        float val = c[t2].f[r] + bia;
        if (v == tgt[t2][r]) rt[t2][r] = fmaxf(rt[t2][r], val);
        float m2 = val;
        for (int msk = 8; msk; msk >>= 1) m2 = fmaxf(m2, __shfl_xor(m2, msk, 32));
        float nm = fmaxf(rm[t2][r], m2);
        float e = __expf(val - nm);
        float es = e;
        for (int msk = 8; msk; msk >>= 1) es += __shfl_xor(es, msk, 32);
        rs[t2][r] = rs[t2][r] * __expf(rm[t2][r] - nm) + es;
        rm[t2][r] = nm;
      }
  }
  for (int t2 = 0; t2 < 2; ++t2)
    for (int r = 0; r < 8; ++r) {
      for (int msk = 8; msk; msk >>= 1)
        rt[t2][r] = fmaxf(rt[t2][r], __shfl_xor(rt[t2][r], msk, 32));
      if (lm == 0) {
        int row = t2 * 16 + r + half8;
        wmaxS[wv][row] = rm[t2][r]; wsumS[wv][row] = rs[t2][r]; wtgtS[wv][row] = rt[t2][r];
      }
    }
  __syncthreads();
  if (tid < 32) {
    float M = -1e30f, S = 0.f, T = -1e30f;
    for (int ww = 0; ww < 4; ++ww) M = fmaxf(M, wmaxS[ww][tid]);
    for (int ww = 0; ww < 4; ++ww) S += wsumS[ww][tid] * __expf(wmaxS[ww][tid] - M);
    for (int ww = 0; ww < 4; ++ww) T = fmaxf(T, wtgtS[ww][tid]);
    loss[rb + tid] = (logf(S) + M) - T;
  }
}

// ---------------- launcher ----------------
extern "C" void kernel_launch(void* const* d_in, const int* in_sizes, int n_in,
                              void* d_out, int out_size, void* d_ws, size_t ws_size,
                              hipStream_t stream)
{
  const int*   dec_inp = (const int*)  d_in[0];
  const int*   target  = (const int*)  d_in[1];
  const float* emb     = (const float*)d_in[2];
  const float* obias   = (const float*)d_in[3];
  const float* Wqkv    = (const float*)d_in[4];
  const float* Wo      = (const float*)d_in[5];
  const float* ln1g    = (const float*)d_in[6];
  const float* ln1b    = (const float*)d_in[7];
  const float* W1      = (const float*)d_in[8];
  const float* W2      = (const float*)d_in[9];
  const float* ln2g    = (const float*)d_in[10];
  const float* ln2b    = (const float*)d_in[11];
  float* loss = (float*)d_out;

  char* p = (char*)d_ws;
  float*          hf32  = (float*)p;          p += (size_t)MM * DMM * 4;    // 16 MB
  unsigned short* hbf   = (unsigned short*)p; p += (size_t)MM * DMM * 2;    //  8 MB
  unsigned short* qkvbf = (unsigned short*)p; p += (size_t)MM * 3072 * 2;   // 24 MB
  unsigned short* attbf = (unsigned short*)p; p += (size_t)MM * DMM * 2;    //  8 MB
  unsigned short* l1bf  = (unsigned short*)p; p += (size_t)MM * DII * 2;    // 32 MB
  float*          tmpf  = (float*)p;          p += (size_t)MM * DMM * 4;    // 16 MB
  unsigned short* wbuf  = (unsigned short*)p; p += (size_t)DII * DMM * 2;   //  8 MB
  unsigned short* embbf = (unsigned short*)p; p += (size_t)VV * DMM * 2;    // 65.5 MB

  // one-time bf16 embedding copy (tied weights: used by the loss sweep)
  cvt_bf16_kernel<<<(VV * DMM) / 1024, 256, 0, stream>>>(emb, embbf, VV * DMM);
  embed_kernel<<<MM, 256, 0, stream>>>(dec_inp, emb, hf32, hbf);

  for (int i = 0; i < LL; ++i) {
    cvt_bf16_kernel<<<(3072 * DMM) / 1024, 256, 0, stream>>>(
        Wqkv + (size_t)i * 3072 * DMM, wbuf, 3072 * DMM);
    gemm_bf16_kernel<true><<<dim3(3072 / 128, MM / 128), 256, 0, stream>>>(
        hbf, wbuf, qkvbf, MM, 3072, DMM);

    attn_kernel<<<dim3(QL / 64, BB * NHH), 128, 0, stream>>>(qkvbf, attbf);

    cvt_bf16_kernel<<<(DMM * DMM) / 1024, 256, 0, stream>>>(
        Wo + (size_t)i * DMM * DMM, wbuf, DMM * DMM);
    gemm_bf16_kernel<false><<<dim3(DMM / 128, MM / 128), 256, 0, stream>>>(
        attbf, wbuf, tmpf, MM, DMM, DMM);

    ln_kernel<<<MM, 256, 0, stream>>>(hf32, tmpf, ln1g + i * DMM, ln1b + i * DMM, hf32, hbf);

    cvt_bf16_kernel<<<(DII * DMM) / 1024, 256, 0, stream>>>(
        W1 + (size_t)i * DII * DMM, wbuf, DII * DMM);
    gemm_bf16_kernel<true><<<dim3(DII / 128, MM / 128), 256, 0, stream>>>(
        hbf, wbuf, l1bf, MM, DII, DMM);

    cvt_bf16_kernel<<<(DMM * DII) / 1024, 256, 0, stream>>>(
        W2 + (size_t)i * DMM * DII, wbuf, DMM * DII);
    gemm_bf16_kernel<false><<<dim3(DMM / 128, MM / 128), 256, 0, stream>>>(
        l1bf, wbuf, tmpf, MM, DMM, DII);

    ln_kernel<<<MM, 256, 0, stream>>>(hf32, tmpf, ln2g + i * DMM, ln2b + i * DMM, hf32, hbf);
  }
  loss_kernel<<<MM / 32, 128, 0, stream>>>(hbf, embbf, obias, target, loss);
}
